// Attention_34995393528541
// MI455X (gfx1250) — matrix operand order repaired and checked by reading, not compiled, not run on hardware
//
#include <hip/hip_runtime.h>

#define B_BATCH 32
#define SEQ     1024
#define DIM     768
#define BS      (B_BATCH * SEQ)
#define KSTEPS  (DIM / 32)   // 24 bf16-WMMA k-steps over D
#define DSLICE  192          // per-wave d-slice in flash kernel (DIM/4)

typedef __attribute__((ext_vector_type(16))) __bf16 v16bf;
typedef __attribute__((ext_vector_type(8)))  __bf16 v8bf;
typedef __attribute__((ext_vector_type(4)))  __bf16 v4bf;
typedef __attribute__((ext_vector_type(8)))  float  v8f;
typedef __attribute__((ext_vector_type(4)))  float  fx4;

__device__ __forceinline__ v8f wmma_bf16(v16bf a, v16bf b, v8f c) {
  // D(f32 16x16) = A(bf16 16x32) * B(bf16 32x16) + C
  return __builtin_amdgcn_wmma_f32_16x16x32_bf16(false, a, false, b, (short)0, c,
                                                 false, false);
}

// A-fragment helper: lane holds two runs of 8
// K values; callers pass K = half*16 + 0..7 and half*16 + 8..15 (same order as B).
__device__ __forceinline__ v16bf cat8(v8bf lo, v8bf hi) {
  v16bf r;
#pragma unroll
  for (int i = 0; i < 8; ++i) { r[i] = lo[i]; r[8 + i] = hi[i]; }
  return r;
}

// ---------------------------------------------------------------------------
// Kernel 0: one-shot fp32 -> bf16 conversion (used for the W matrices so the
// GEMM hot loop reads bf16 directly: 1x 32B load per B-fragment, no cvts).
// ---------------------------------------------------------------------------
__global__ __launch_bounds__(256) void cvt_f32_bf16_kernel(
    const float* __restrict__ src, __bf16* __restrict__ dst, int n4)
{
  const int i = blockIdx.x * 256 + threadIdx.x;
  if (i < n4) {
    fx4 v = ((const fx4*)src)[i];
    v4bf o;
    o[0] = (__bf16)v[0]; o[1] = (__bf16)v[1];
    o[2] = (__bf16)v[2]; o[3] = (__bf16)v[3];
    ((v4bf*)dst)[i] = o;
  }
}

// ---------------------------------------------------------------------------
// Kernel A: Q/K/V = x*W^T + b, RoPE(Q,K), 1/sqrt(D) folded into Q.
// Q,K stored row-major bf16 [B*S][D]; V stored transposed bf16 [B][D][S].
// Block = 256 threads (8 waves), covers TWO 16-row m-tiles (32 rows of x in
// LDS) x 8 feature tiles. Each W fragment is loaded once (bf16, contiguous
// 32B) and reused for both m-tiles: hot loop = 6 WMMA : 3 global : 4 ds.
// ---------------------------------------------------------------------------
__global__ __launch_bounds__(256) void qkv_rope_kernel(
    const float* __restrict__ x,
    const __bf16* __restrict__ Wqb, const float* __restrict__ bq,
    const __bf16* __restrict__ Wkb, const float* __restrict__ bk,
    const __bf16* __restrict__ Wvb, const float* __restrict__ bv,
    __bf16* __restrict__ Qb, __bf16* __restrict__ Kb, __bf16* __restrict__ Vt)
{
  __shared__ __bf16 Xs[32 * DIM];               // 48 KB: two 16-row x tiles

  const int lane = threadIdx.x & 31;
  const int wave = threadIdx.x >> 5;
  const int half = lane >> 4;
  const int mt0  = blockIdx.x * 2;              // first of two m-tiles
  const int nt   = blockIdx.y * 8 + wave;       // 16-col feature tile (0..47)

  // Stage 32 rows of x into LDS as bf16 (rows contiguous in memory).
  const float* xblk = x + (size_t)mt0 * 16 * DIM;
  for (int i4 = threadIdx.x; i4 < 32 * DIM / 4; i4 += 256) {
    fx4 v = ((const fx4*)xblk)[i4];
    v4bf o;
    o[0] = (__bf16)v[0]; o[1] = (__bf16)v[1];
    o[2] = (__bf16)v[2]; o[3] = (__bf16)v[3];
    *(v4bf*)(Xs + i4 * 4) = o;
  }
  __syncthreads();

  const int e = nt * 16 + (lane & 15);          // output feature = W row
  const __bf16* wqr = Wqb + (size_t)e * DIM;
  const __bf16* wkr = Wkb + (size_t)e * DIM;
  const __bf16* wvr = Wvb + (size_t)e * DIM;

  v8f cq[2] = {{}, {}}, ck[2] = {{}, {}}, cv[2] = {{}, {}};
#pragma unroll 4
  for (int kt = 0; kt < KSTEPS; ++kt) {
    const int ka = kt * 32 + half * 16;
    v16bf a0 = cat8(*(const v8bf*)(Xs + (lane & 15) * DIM + ka),
                    *(const v8bf*)(Xs + (lane & 15) * DIM + ka + 8));
    v16bf a1 = cat8(*(const v8bf*)(Xs + (16 + (lane & 15)) * DIM + ka),
                    *(const v8bf*)(Xs + (16 + (lane & 15)) * DIM + ka + 8));
    const int kb = kt * 32 + half * 16;         // B: lane = N + K-half
    v16bf fq = *(const v16bf*)(wqr + kb);       // 32B contiguous bf16
    v16bf fk = *(const v16bf*)(wkr + kb);
    v16bf fv = *(const v16bf*)(wvr + kb);
    cq[0] = wmma_bf16(a0, fq, cq[0]);  cq[1] = wmma_bf16(a1, fq, cq[1]);
    ck[0] = wmma_bf16(a0, fk, ck[0]);  ck[1] = wmma_bf16(a1, fk, ck[1]);
    cv[0] = wmma_bf16(a0, fv, cv[0]);  cv[1] = wmma_bf16(a1, fv, cv[1]);
  }

  const float bqe = bq[e], bke = bk[e], bve = bv[e];
  const float invsq = 0.03608439182435161f;     // 1/sqrt(768)
  // inv_freq = 10000^(-2*(e/2)/D); same for both lanes of an (even,odd) pair
  const float freq = __expf(-((float)((e >> 1) * 2) / (float)DIM) * 9.210340371976184f);
  const bool  odd  = (e & 1) != 0;

#pragma unroll
  for (int ms = 0; ms < 2; ++ms) {
    const int mt = mt0 + ms;
    const int bb = mt >> 6;                     // batch index (S/16 == 64)
    const int s0 = (mt & 63) * 16;              // tile's base sequence position
    v8bf vv;
#pragma unroll
    for (int r = 0; r < 8; ++r) {               // C row = r + 8*half, col = e
      const int   srow = s0 + r + 8 * half;
      const int   grow = bb * SEQ + srow;
      const float ang  = (float)srow * freq;
      const float cs = __cosf(ang), sn = __sinf(ang);

      float qv = cq[ms][r] + bqe;
      float qp = __shfl_xor(qv, 1, 32);         // RoPE partner (adjacent feature)
      float qo = odd ? (qp * sn + qv * cs) : (qv * cs - qp * sn);
      Qb[(size_t)grow * DIM + e] = (__bf16)(qo * invsq);

      float kv = ck[ms][r] + bke;
      float kp = __shfl_xor(kv, 1, 32);
      float ko = odd ? (kp * sn + kv * cs) : (kv * cs - kp * sn);
      Kb[(size_t)grow * DIM + e] = (__bf16)ko;

      vv[r] = (__bf16)(cv[ms][r] + bve);
    }
    // V transposed: Vt[b][d][s]; the 8 rows of this lane are consecutive s.
    *(v8bf*)(Vt + ((size_t)bb * DIM + e) * SEQ + s0 + 8 * half) = vv;
  }
}

// ---------------------------------------------------------------------------
// Kernel B: flash attention, fully balanced.
// Block = 256 threads (8 waves) = 2 q-tiles x 4 waves. Each wave owns a
// 192-wide d-slice: it computes the partial scores for that slice (12 WMMAs)
// and the P@V output for that slice (12 WMMAs). Slice 0 reduces partials and
// runs the online softmax. Q tiles staged once in LDS.
// ---------------------------------------------------------------------------
__global__ __launch_bounds__(256) void flash_attn_kernel(
    const __bf16* __restrict__ Qb, const __bf16* __restrict__ Kb,
    const __bf16* __restrict__ Vt, float* __restrict__ out)
{
  __shared__ __bf16 Qs[2][16 * DIM];     // 48 KB: both q-tiles of this block
  __shared__ float  Pa[2][3][32 * 16];   // 12 KB: partials from slices 1..3
  __shared__ __bf16 Ps[2][16 * 32];      // 2 KB: probability tiles (bf16)
  __shared__ float  alphaS[2][16];       // per-row rescale factor
  __shared__ float  linvS[2][16];        // per-row 1/l

  const int lane = threadIdx.x & 31;
  const int wave = threadIdx.x >> 5;
  const int half = lane >> 4;
  const int qi   = wave >> 2;            // which q-tile in block (0..1)
  const int ws   = wave & 3;             // d-slice index (0..3)
  const int b    = blockIdx.x;
  const int qt   = blockIdx.y * 2 + qi;  // q tile (0..63)
  const size_t qrow0 = (size_t)b * SEQ + qt * 16;

  // Stage both 16x768 Q tiles (rows contiguous across the two tiles).
  {
    const v8bf* src = (const v8bf*)(Qb + ((size_t)b * SEQ + blockIdx.y * 32) * DIM);
    for (int i = threadIdx.x; i < 2 * 16 * DIM / 8; i += 256)
      ((v8bf*)Qs)[i] = src[i];
  }
  __syncthreads();

  v8f acc[12];
#pragma unroll
  for (int t = 0; t < 12; ++t) acc[t] = (v8f){};
  float m_i[8], l_i[8];
#pragma unroll
  for (int r = 0; r < 8; ++r) { m_i[r] = -1e30f; l_i[r] = 0.f; }

  for (int kc = 0; kc < SEQ / 32; ++kc) {
    const int key0 = kc * 32;
    const __bf16* Krow = Kb + ((size_t)b * SEQ + key0) * DIM + ws * DSLICE;

    // Cover next chunk's K/V latency (global_prefetch_b8).
    if (kc + 1 < SEQ / 32) {
      __builtin_prefetch(Krow + (size_t)(32 + (lane & 15)) * DIM, 0, 1);
      __builtin_prefetch(Vt + ((size_t)b * DIM + ws * DSLICE + (lane & 15)) * SEQ
                            + key0 + 32, 0, 1);
    }

    // Partial scores over this wave's 192-wide d-slice: 6 k-steps x 2 key-tiles
    v8f c0 = {}, c1 = {};
#pragma unroll
    for (int kt = 0; kt < DSLICE / 32; ++kt) {
      const int ka = ws * DSLICE + kt * 32 + half * 16;
      v16bf af = cat8(*(const v8bf*)(&Qs[qi][(lane & 15) * DIM + ka]),
                      *(const v8bf*)(&Qs[qi][(lane & 15) * DIM + ka + 8]));
      const int kb = kt * 32 + half * 16;
      v16bf b0 = *(const v16bf*)(Krow + (size_t)(lane & 15) * DIM + kb);
      v16bf b1 = *(const v16bf*)(Krow + (size_t)(16 + (lane & 15)) * DIM + kb);
      c0 = wmma_bf16(af, b0, c0);
      c1 = wmma_bf16(af, b1, c1);
    }

    if (ws != 0) {                       // export partials to the reducer wave
      float* pb = &Pa[qi][ws - 1][lane * 16];
      *(fx4*)(pb + 0)  = (fx4){c0[0], c0[1], c0[2], c0[3]};
      *(fx4*)(pb + 4)  = (fx4){c0[4], c0[5], c0[6], c0[7]};
      *(fx4*)(pb + 8)  = (fx4){c1[0], c1[1], c1[2], c1[3]};
      *(fx4*)(pb + 12) = (fx4){c1[4], c1[5], c1[6], c1[7]};
    }
    __syncthreads();

    if (ws == 0) {
      // Reduce the three partial tiles into c0/c1.
#pragma unroll
      for (int w = 0; w < 3; ++w) {
        const float* pb = &Pa[qi][w][lane * 16];
        fx4 a0 = *(const fx4*)(pb + 0);
        fx4 a1 = *(const fx4*)(pb + 4);
        fx4 a2 = *(const fx4*)(pb + 8);
        fx4 a3 = *(const fx4*)(pb + 12);
#pragma unroll
        for (int i = 0; i < 4; ++i) {
          c0[i] += a0[i]; c0[4 + i] += a1[i];
          c1[i] += a2[i]; c1[4 + i] += a3[i];
        }
      }
      // Online softmax: row reductions across the 16 lanes of each half.
#pragma unroll
      for (int r = 0; r < 8; ++r) {
        float t = fmaxf(c0[r], c1[r]);
        t = fmaxf(t, __shfl_xor(t, 1, 32));
        t = fmaxf(t, __shfl_xor(t, 2, 32));
        t = fmaxf(t, __shfl_xor(t, 4, 32));
        t = fmaxf(t, __shfl_xor(t, 8, 32));
        const float mnew  = fmaxf(m_i[r], t);
        const float alpha = __expf(m_i[r] - mnew);
        const float p0 = __expf(c0[r] - mnew);
        const float p1 = __expf(c1[r] - mnew);
        float rs = p0 + p1;
        rs += __shfl_xor(rs, 1, 32);
        rs += __shfl_xor(rs, 2, 32);
        rs += __shfl_xor(rs, 4, 32);
        rs += __shfl_xor(rs, 8, 32);
        l_i[r] = l_i[r] * alpha + rs;
        m_i[r] = mnew;
        const int rowp = r + 8 * half;
        if ((lane & 15) == 0) alphaS[qi][rowp] = alpha;
        Ps[qi][rowp * 32 + (lane & 15)]      = (__bf16)p0;
        Ps[qi][rowp * 32 + 16 + (lane & 15)] = (__bf16)p1;
      }
    }
    __syncthreads();

    // All waves: acc = acc*alpha + P(16x32) @ V(32 x 192-slice)
    v16bf pf = cat8(*(const v8bf*)(&Ps[qi][(lane & 15) * 32 + half * 16]),
                    *(const v8bf*)(&Ps[qi][(lane & 15) * 32 + half * 16 + 8]));
#pragma unroll
    for (int r = 0; r < 8; ++r) {
      const float a = alphaS[qi][r + 8 * half];
#pragma unroll
      for (int t = 0; t < 12; ++t) acc[t][r] *= a;
    }
#pragma unroll
    for (int t = 0; t < 12; ++t) {
      const int d0 = ws * DSLICE + t * 16;
      const __bf16* vp =
          Vt + ((size_t)b * DIM + d0 + (lane & 15)) * SEQ + key0 + half * 16;
      v16bf bv = *(const v16bf*)vp;      // 32B contiguous (V transposed)
      acc[t] = wmma_bf16(pf, bv, acc[t]);
    }
    __syncthreads();                     // Pa/Ps/alphaS reused next chunk
  }

  if (ws == 0 && (lane & 15) == 0) {
#pragma unroll
    for (int r = 0; r < 8; ++r) linvS[qi][r + 8 * half] = 1.0f / l_i[r];
  }
  __syncthreads();

#pragma unroll
  for (int r = 0; r < 8; ++r) {
    const float li = linvS[qi][r + 8 * half];
    float* op = out + (qrow0 + r + 8 * half) * DIM + ws * DSLICE + (lane & 15);
#pragma unroll
    for (int t = 0; t < 12; ++t) op[t * 16] = acc[t][r] * li;
  }
}

// ---------------------------------------------------------------------------
extern "C" void kernel_launch(void* const* d_in, const int* in_sizes, int n_in,
                              void* d_out, int out_size, void* d_ws, size_t ws_size,
                              hipStream_t stream) {
  (void)in_sizes; (void)n_in; (void)out_size; (void)ws_size;
  const float* x  = (const float*)d_in[0];
  const float* Wq = (const float*)d_in[1];
  const float* bq = (const float*)d_in[2];
  const float* Wk = (const float*)d_in[3];
  const float* bk = (const float*)d_in[4];
  const float* Wv = (const float*)d_in[5];
  const float* bv = (const float*)d_in[6];
  float* out = (float*)d_out;

  // Workspace: Q, K row-major bf16 + V transposed bf16 (3 x 48 MB) followed
  // by the three bf16 weight matrices (3 x 1.125 MB). Total ~147.4 MB.
  const size_t mat  = (size_t)BS * DIM * sizeof(__bf16);
  const size_t wmat = (size_t)DIM * DIM * sizeof(__bf16);
  __bf16* Qb  = (__bf16*)d_ws;
  __bf16* Kb  = (__bf16*)((char*)d_ws + mat);
  __bf16* Vt  = (__bf16*)((char*)d_ws + 2 * mat);
  __bf16* Wqb = (__bf16*)((char*)d_ws + 3 * mat);
  __bf16* Wkb = (__bf16*)((char*)d_ws + 3 * mat + wmat);
  __bf16* Wvb = (__bf16*)((char*)d_ws + 3 * mat + 2 * wmat);

  // One-shot weight conversion to bf16 (stream-ordered before the GEMM).
  const int wn4 = DIM * DIM / 4;
  const int cb  = (wn4 + 255) / 256;
  cvt_f32_bf16_kernel<<<cb, 256, 0, stream>>>(Wq, Wqb, wn4);
  cvt_f32_bf16_kernel<<<cb, 256, 0, stream>>>(Wk, Wkb, wn4);
  cvt_f32_bf16_kernel<<<cb, 256, 0, stream>>>(Wv, Wvb, wn4);

  dim3 gA(BS / 32, DIM / (16 * 8));   // 1024 x 6 blocks, 2 m-tiles per block
  qkv_rope_kernel<<<gA, 256, 0, stream>>>(x, Wqb, bq, Wkb, bk, Wvb, bv,
                                          Qb, Kb, Vt);

  dim3 gB(B_BATCH, SEQ / 32);         // 32 x 32 blocks, 2 q-tiles per block
  flash_attn_kernel<<<gB, 256, 0, stream>>>(Qb, Kb, Vt, out);
}
